// Pooler_4398046511881
// MI455X (gfx1250) — matrix operand order repaired and checked by reading, not compiled, not run on hardware
//
#include <hip/hip_runtime.h>
#include <math.h>

typedef __attribute__((ext_vector_type(16))) __bf16 v16bf;
typedef __attribute__((ext_vector_type(8)))  __bf16 v8bf;
typedef __attribute__((ext_vector_type(8)))  float  v8f;

#define DIM   256
#define H1    128
#define H2    64
#define WAVES 4   // waves per MLP block; 4 * 16 = 64 rows / block

// ---------------------------------------------------------------------------
// Prep: convert W[K][N] (f32 row-major) into bf16 WMMA B-fragments:
//   frag f = nt*(K/32)+kt ; lane 0..31 owns 16 contiguous bf16:
//   element i of lane = W[(kt*32 + (lane>>4)*8 + (i&7) + (i>>3)*16)*N + nt*16 + (lane&15)]
// ---------------------------------------------------------------------------
__global__ __launch_bounds__(256)
void swizzle_bfrag_kernel(const float* __restrict__ W, int K, int N,
                          __bf16* __restrict__ outfrag) {
    const int idx = blockIdx.x * 256 + threadIdx.x;
    if (idx >= K * N) return;
    const int f    = idx >> 9;
    const int r    = idx & 511;
    const int lane = r >> 4;
    const int i    = r & 15;
    const int ktiles = K >> 5;
    const int nt = f / ktiles;
    const int kt = f - nt * ktiles;
    const int Kidx = kt * 32 + ((lane >> 4) << 3) + (i & 7) + ((i >> 3) << 4);
    const int Nidx = nt * 16 + (lane & 15);
    outfrag[idx] = (__bf16)W[(size_t)Kidx * N + Nidx];
}

__device__ __forceinline__ v16bf load_frag32(const __bf16* p) {
    v8bf lo = *(const v8bf*)p;
    v8bf hi = *(const v8bf*)(p + 8);
    return __builtin_shufflevector(lo, hi,
        0, 1, 2, 3, 4, 5, 6, 7, 8, 9, 10, 11, 12, 13, 14, 15);
}

// ---------------------------------------------------------------------------
// Kernel 1: per-row gating MLP, one wave per 16-row tile, bf16 WMMA.
// B-fragments are preloaded per n-tile so loads overlap the WMMA chain.
// ---------------------------------------------------------------------------
__global__ __launch_bounds__(WAVES * 32)
void mlp_logits_kernel(const float* __restrict__ src, int nrows,
                       const __bf16* __restrict__ fragW1,
                       const __bf16* __restrict__ fragW2,
                       const float* __restrict__ b1, const float* __restrict__ b2,
                       const float* __restrict__ W3, const float* __restrict__ b3,
                       float* __restrict__ logits) {
    __shared__ __align__(16) __bf16 sW2f[H1 * H2];          // 16 KB fragment layout
    __shared__ float  sb1[H1];
    __shared__ float  sb2[H2];
    __shared__ float  sW3[H2];
    __shared__ float  sb3s;
    __shared__ __align__(16) __bf16 sH1[WAVES][16 * H1];    // 4 KB per wave

    const int tid = threadIdx.x;
    {
        const uint4* sv = (const uint4*)fragW2;
        uint4*       dv = (uint4*)sW2f;
        for (int i = tid; i < (H1 * H2) / 8; i += WAVES * 32) dv[i] = sv[i];
        for (int i = tid; i < H1; i += WAVES * 32) sb1[i] = b1[i];
        for (int i = tid; i < H2; i += WAVES * 32) { sb2[i] = b2[i]; sW3[i] = W3[i]; }
        if (tid == 0) sb3s = b3[0];
    }
    __syncthreads();

    const int wave = tid >> 5;
    const int lane = tid & 31;
    const int tiles = (nrows + 15) >> 4;
    const long tile = (long)blockIdx.x * WAVES + wave;
    if (tile >= tiles) return;                 // wave-uniform; no more block syncs

    const int row0 = (int)(tile << 4);
    const int m    = lane & 15;
    const int half = lane >> 4;
    const int kb   = half << 3;
    const float cb3 = sb3s;

    // ---- A fragments: 16 rows x 256 cols as 8 (16x32 bf16) fragments ----
    const int  arow   = row0 + m;
    const bool rvalid = arow < nrows;
    const float* srow = src + (size_t)arow * DIM;

    v16bf afrag[8];
#pragma unroll
    for (int kt = 0; kt < 8; ++kt) {
        v16bf a;
#pragma unroll
        for (int g = 0; g < 2; ++g) {
            const int K0 = kt * 32 + g * 16 + kb;
            float4 x = rvalid ? *(const float4*)(srow + K0)     : make_float4(0.f, 0.f, 0.f, 0.f);
            float4 y = rvalid ? *(const float4*)(srow + K0 + 4) : make_float4(0.f, 0.f, 0.f, 0.f);
            a[g * 8 + 0] = (__bf16)x.x; a[g * 8 + 1] = (__bf16)x.y;
            a[g * 8 + 2] = (__bf16)x.z; a[g * 8 + 3] = (__bf16)x.w;
            a[g * 8 + 4] = (__bf16)y.x; a[g * 8 + 5] = (__bf16)y.y;
            a[g * 8 + 6] = (__bf16)y.z; a[g * 8 + 7] = (__bf16)y.w;
        }
        afrag[kt] = a;
    }

    // ---- layer 1: h1 = relu(A @ W1 + b1) -> LDS (row-major 16x128 bf16) ----
#pragma unroll
    for (int nt = 0; nt < 8; ++nt) {
        const int n0 = nt * 16;
        const float bias = sb1[n0 + m];

        // preload all 8 B fragments of this n-tile: 16 b128 loads in flight
        v16bf bfr[8];
#pragma unroll
        for (int kt = 0; kt < 8; ++kt)
            bfr[kt] = load_frag32(fragW1 + ((size_t)(nt * 8 + kt) * 512 + lane * 16));

        v8f acc;
#pragma unroll
        for (int v = 0; v < 8; ++v) acc[v] = bias;
#pragma unroll
        for (int kt = 0; kt < 8; ++kt)
            acc = __builtin_amdgcn_wmma_f32_16x16x32_bf16(
                false, afrag[kt], false, bfr[kt], (short)0, acc, false, false);

#pragma unroll
        for (int v = 0; v < 8; ++v) {
            const float r = acc[v] > 0.f ? acc[v] : 0.f;
            const int M = v + (half << 3);
            sH1[wave][M * H1 + n0 + m] = (__bf16)r;
        }
    }

    // wave-local LDS RAW fence (cross-lane store -> load)
    asm volatile("s_wait_dscnt 0" ::: "memory");

    // ---- layer 2 + layer 3 fused ----
    float part[8];
#pragma unroll
    for (int v = 0; v < 8; ++v) part[v] = 0.f;

#pragma unroll
    for (int nt = 0; nt < 4; ++nt) {
        const int n0 = nt * 16;
        const float bias = sb2[n0 + m];
        const float w3v  = sW3[n0 + m];

        v16bf a2[4], b2f[4];
#pragma unroll
        for (int kt = 0; kt < 4; ++kt) {
            const __bf16* ph = &sH1[wave][m * H1 + kt * 32 + kb];
            v8bf alo = *(const v8bf*)ph;
            v8bf ahi = *(const v8bf*)(ph + 16);
            a2[kt] = __builtin_shufflevector(alo, ahi,
                0, 1, 2, 3, 4, 5, 6, 7, 8, 9, 10, 11, 12, 13, 14, 15);
            b2f[kt] = load_frag32(&sW2f[(nt * 4 + kt) * 512 + lane * 16]);
        }

        v8f acc;
#pragma unroll
        for (int v = 0; v < 8; ++v) acc[v] = bias;
#pragma unroll
        for (int kt = 0; kt < 4; ++kt)
            acc = __builtin_amdgcn_wmma_f32_16x16x32_bf16(
                false, a2[kt], false, b2f[kt], (short)0, acc, false, false);

#pragma unroll
        for (int v = 0; v < 8; ++v) {
            const float r = acc[v] > 0.f ? acc[v] : 0.f;
            part[v] += r * w3v;
        }
    }

    // reduce across the 16 lanes of each half (xor masks stay within a half)
#pragma unroll
    for (int off = 1; off < 16; off <<= 1) {
#pragma unroll
        for (int v = 0; v < 8; ++v) part[v] += __shfl_xor(part[v], off, 32);
    }

    if (m == 0) {
#pragma unroll
        for (int v = 0; v < 8; ++v) {
            const int row = row0 + v + (half << 3);
            if (row < nrows) logits[row] = part[v] + cb3;
        }
    }
}

// ---------------------------------------------------------------------------
// sorted-index segment helpers
// ---------------------------------------------------------------------------
__device__ __forceinline__ int lower_bound_i(const int* __restrict__ idx, int n, int key) {
    int lo = 0, hi = n;
    while (lo < hi) {
        int mid = (lo + hi) >> 1;
        if (idx[mid] < key) lo = mid + 1; else hi = mid;
    }
    return lo;
}

__global__ __launch_bounds__(256)
void seg_softmax_kernel(const float* __restrict__ logits, const int* __restrict__ index,
                        int n, float* __restrict__ w) {
    const int g = blockIdx.x;
    const int tid = threadIdx.x;
    const int start = lower_bound_i(index, n, g);
    const int end   = lower_bound_i(index, n, g + 1);
    __shared__ float red[256];

    float mx = -3.402823466e38f;
    for (int r = start + tid; r < end; r += 256) mx = fmaxf(mx, logits[r]);
    red[tid] = mx; __syncthreads();
    for (int s = 128; s > 0; s >>= 1) { if (tid < s) red[tid] = fmaxf(red[tid], red[tid + s]); __syncthreads(); }
    mx = red[0]; __syncthreads();

    float sum = 0.f;
    for (int r = start + tid; r < end; r += 256) sum += __expf(logits[r] - mx);
    red[tid] = sum; __syncthreads();
    for (int s = 128; s > 0; s >>= 1) { if (tid < s) red[tid] += red[tid + s]; __syncthreads(); }
    const float inv = 1.0f / red[0];

    for (int r = start + tid; r < end; r += 256) w[r] = __expf(logits[r] - mx) * inv;
}

// Kernel 3: weighted segment-sum; thread-per-feature, coalesced streaming.
// HBM-roofline phase: 2-way unroll + prefetch for more loads in flight.
__global__ __launch_bounds__(256)
void seg_pool_kernel(const float* __restrict__ src, const float* __restrict__ w,
                     const int* __restrict__ index, int n, float* __restrict__ out) {
    const int g = blockIdx.x;
    const int f = threadIdx.x;                 // DIM == blockDim.x == 256
    const int start = lower_bound_i(index, n, g);
    const int end   = lower_bound_i(index, n, g + 1);
    float acc0 = 0.f, acc1 = 0.f;
    int r = start;
    for (; r + 1 < end; r += 2) {
        if (r + 8 < end)
            __builtin_prefetch(src + (size_t)(r + 8) * DIM + f, 0, 3);
        acc0 += w[r]     * src[(size_t)r * DIM + f];
        acc1 += w[r + 1] * src[(size_t)(r + 1) * DIM + f];
    }
    if (r < end) acc0 += w[r] * src[(size_t)r * DIM + f];
    out[(size_t)g * DIM + f] = acc0 + acc1;    // writes 0 for empty segments
}

// ---------------------------------------------------------------------------
extern "C" void kernel_launch(void* const* d_in, const int* in_sizes, int n_in,
                              void* d_out, int out_size, void* d_ws, size_t ws_size,
                              hipStream_t stream) {
    const float* emb_nodes = (const float*)d_in[0];
    const float* emb_edges = (const float*)d_in[1];
    const int*   node_index = (const int*)d_in[2];
    const int*   edge_index = (const int*)d_in[3];
    const float* Wn1 = (const float*)d_in[4];  const float* bn1 = (const float*)d_in[5];
    const float* Wn2 = (const float*)d_in[6];  const float* bn2 = (const float*)d_in[7];
    const float* Wn3 = (const float*)d_in[8];  const float* bn3 = (const float*)d_in[9];
    const float* We1 = (const float*)d_in[10]; const float* be1 = (const float*)d_in[11];
    const float* We2 = (const float*)d_in[12]; const float* be2 = (const float*)d_in[13];
    const float* We3 = (const float*)d_in[14]; const float* be3 = (const float*)d_in[15];

    const int Nn = in_sizes[0] / DIM;
    const int Ne = in_sizes[1] / DIM;
    const int G  = out_size / (2 * DIM);       // num_graphs

    float* out = (float*)d_out;
    float* ws  = (float*)d_ws;
    float* logits_n = ws;
    float* w_n      = logits_n + Nn;
    float* logits_e = w_n + Nn;
    float* w_e      = logits_e + Ne;
    __bf16* fW1n = (__bf16*)(w_e + Ne);        // 32768 bf16 (64 KB)
    __bf16* fW2n = fW1n + DIM * H1;            //  8192 bf16 (16 KB)
    __bf16* fW1e = fW2n + H1 * H2;
    __bf16* fW2e = fW1e + DIM * H1;

    swizzle_bfrag_kernel<<<dim3((DIM * H1 + 255) / 256), dim3(256), 0, stream>>>(Wn1, DIM, H1, fW1n);
    swizzle_bfrag_kernel<<<dim3((H1 * H2 + 255) / 256),  dim3(256), 0, stream>>>(Wn2, H1, H2, fW2n);
    swizzle_bfrag_kernel<<<dim3((DIM * H1 + 255) / 256), dim3(256), 0, stream>>>(We1, DIM, H1, fW1e);
    swizzle_bfrag_kernel<<<dim3((H1 * H2 + 255) / 256),  dim3(256), 0, stream>>>(We2, H1, H2, fW2e);

    const int rows_per_block = WAVES * 16;
    dim3 blkMlp(WAVES * 32);
    dim3 grdN((Nn + rows_per_block - 1) / rows_per_block);
    dim3 grdE((Ne + rows_per_block - 1) / rows_per_block);

    mlp_logits_kernel<<<grdN, blkMlp, 0, stream>>>(emb_nodes, Nn, fW1n, fW2n, bn1, bn2, Wn3, bn3, logits_n);
    mlp_logits_kernel<<<grdE, blkMlp, 0, stream>>>(emb_edges, Ne, fW1e, fW2e, be1, be2, We3, be3, logits_e);

    seg_softmax_kernel<<<dim3(G), dim3(256), 0, stream>>>(logits_n, node_index, Nn, w_n);
    seg_softmax_kernel<<<dim3(G), dim3(256), 0, stream>>>(logits_e, edge_index, Ne, w_e);

    seg_pool_kernel<<<dim3(G), dim3(256), 0, stream>>>(emb_nodes, w_n, node_index, Nn, out);
    seg_pool_kernel<<<dim3(G), dim3(256), 0, stream>>>(emb_edges, w_e, edge_index, Ne, out + (size_t)G * DIM);
}
